// Transformer_65644280152463
// MI455X (gfx1250) — matrix operand order repaired and checked by reading, not compiled, not run on hardware
//
#include <hip/hip_runtime.h>

// ---------------------------------------------------------------------------
// Transformer (enc-dec with buggy MHA) forward for MI455X / gfx1250.
//
// DCE faithful to the reference under jit: encoder branch and wk/wv
// projections are dead -> skipped.  Remaining: 6 decoder layers + final LN +
// 768x32000 vocab projection, ~136 GFLOP of GEMMs.  Live weights ~57M params
// -> ~115MB bf16, fits the 192MB L2; matrix-core bound -> bf16 WMMA with f32
// accumulate.  GEMM: 128x128 block tile, 4 waves x (64x64 macro-tile) = 16
// v_wmma per K-step per wave at a 1:1 ds_load_b128:wmma ratio, double-buffered
// LDS fed by GLOBAL_LOAD_ASYNC_TO_LDS_B128 (ASYNCcnt) when available.
// ---------------------------------------------------------------------------

typedef __bf16 bf16;
typedef __attribute__((ext_vector_type(16))) __bf16 bf16x16;
typedef __attribute__((ext_vector_type(8)))  __bf16 bf16x8;
typedef __attribute__((ext_vector_type(8)))  float  f32x8;

#define TOK   1024      // B*L
#define NDIM  768
#define NFF   2048
#define NHEAD 12
#define DH    64
#define LSEQ  512
#define BH    24        // B*NHEAD
#define VOCAB 32000

// ----------------------- async global->LDS plumbing ------------------------
#if defined(__has_builtin)
#  if __has_builtin(__builtin_amdgcn_global_load_async_to_lds_b128)
#    define USE_ASYNC_LDS 1
#  endif
#  if __has_builtin(__builtin_amdgcn_s_wait_asynccnt)
#    define HAVE_WAIT_ASYNC_BUILTIN 1
#  endif
#endif

__device__ __forceinline__ void wait_async0() {
#if defined(USE_ASYNC_LDS)
#  if defined(HAVE_WAIT_ASYNC_BUILTIN)
    __builtin_amdgcn_s_wait_asynccnt(0);
#  else
    asm volatile("s_wait_asynccnt 0x0" ::: "memory");
#  endif
#endif
}

#if defined(USE_ASYNC_LDS)
// builtin signature (from hipcc diagnostic): param 1 is a non-const
// v4i (int __vector_size__(16)) pointer in the global (__device__/AS1)
// address space; LDS side is the AS3 analogue.  (src global, dst lds, imm
// offset, imm cpol) -- the differing address spaces type-check the order.
typedef __attribute__((__vector_size__(16))) int v4i_;
__device__ __forceinline__ void async_cp16(const bf16* g, bf16* l) {
    __builtin_amdgcn_global_load_async_to_lds_b128(
        (__attribute__((address_space(1))) v4i_*)g,
        (__attribute__((address_space(3))) v4i_*)l,
        0, 0);
}
#endif

// -------------------------- f32 -> bf16 convert ----------------------------
__global__ void cvt_f32_bf16(const float* __restrict__ src, bf16* __restrict__ dst, int n) {
    int i = blockIdx.x * 256 + threadIdx.x;
    if (i < n) dst[i] = (bf16)src[i];
}

__global__ void copy_f32(const float* __restrict__ s, float* __restrict__ d, int n) {
    int i = blockIdx.x * 256 + threadIdx.x;
    if (i < n) d[i] = s[i];
}

// ------------------------------- LayerNorm ---------------------------------
__global__ __launch_bounds__(128)
void layernorm768_bf16(const float* __restrict__ x, const float* __restrict__ w,
                       const float* __restrict__ bia, bf16* __restrict__ y) {
    __shared__ float red[128];
    const int row = blockIdx.x, tid = threadIdx.x;
    const float* xr = x + (size_t)row * NDIM;
    float v[6]; float s = 0.f;
#pragma unroll
    for (int i = 0; i < 6; ++i) { v[i] = xr[tid + 128 * i]; s += v[i]; }
    red[tid] = s; __syncthreads();
    for (int st = 64; st > 0; st >>= 1) { if (tid < st) red[tid] += red[tid + st]; __syncthreads(); }
    const float mu = red[0] * (1.f / NDIM);
    __syncthreads();
    float vs = 0.f;
#pragma unroll
    for (int i = 0; i < 6; ++i) { float d = v[i] - mu; vs += d * d; }
    red[tid] = vs; __syncthreads();
    for (int st = 64; st > 0; st >>= 1) { if (tid < st) red[tid] += red[tid + st]; __syncthreads(); }
    const float rinv = rsqrtf(red[0] * (1.f / NDIM) + 1e-5f);
    bf16* yr = y + (size_t)row * NDIM;
#pragma unroll
    for (int i = 0; i < 6; ++i) {
        int c = tid + 128 * i;
        yr[c] = (bf16)((v[i] - mu) * rinv * w[c] + bia[c]);
    }
}

// ------------------------------- Softmax -----------------------------------
__global__ __launch_bounds__(256)
void softmax512_bf16(const float* __restrict__ S, const float* __restrict__ mask,
                     bf16* __restrict__ P) {
    __shared__ float red[256];
    const int bh = blockIdx.y, row = blockIdx.x, tid = threadIdx.x;
    const float* sr = S + ((size_t)bh * LSEQ + row) * LSEQ;
    float a = sr[tid], b = sr[tid + 256];
    if (mask) { a += mask[(size_t)row * LSEQ + tid]; b += mask[(size_t)row * LSEQ + tid + 256]; }
    float m = fmaxf(a, b);
    red[tid] = m; __syncthreads();
    for (int st = 128; st > 0; st >>= 1) { if (tid < st) red[tid] = fmaxf(red[tid], red[tid + st]); __syncthreads(); }
    m = red[0]; __syncthreads();
    float ea = __expf(a - m), eb = __expf(b - m);
    red[tid] = ea + eb; __syncthreads();
    for (int st = 128; st > 0; st >>= 1) { if (tid < st) red[tid] += red[tid + st]; __syncthreads(); }
    const float inv = 1.f / red[0];
    bf16* pr = P + ((size_t)bh * LSEQ + row) * LSEQ;
    pr[tid]       = (bf16)(ea * inv);
    pr[tid + 256] = (bf16)(eb * inv);
}

// --------------------------- head split / merge ----------------------------
__global__ void split_heads(const bf16* __restrict__ q, bf16* __restrict__ qh,
                            bf16* __restrict__ qhT) {
    int i = blockIdx.x * 256 + threadIdx.x;
    if (i >= TOK * NDIM) return;
    int tok = i / NDIM, c = i % NDIM;
    int b = tok >> 9, l = tok & 511;
    int h = c >> 6,  d = c & 63;
    int bh = b * NHEAD + h;
    bf16 v = q[i];
    qh [((size_t)bh * LSEQ + l) * DH + d] = v;
    qhT[((size_t)bh * DH   + d) * LSEQ + l] = v;
}

__global__ void merge_heads(const bf16* __restrict__ O, bf16* __restrict__ cat) {
    int i = blockIdx.x * 256 + threadIdx.x;
    if (i >= TOK * NDIM) return;
    int tok = i / NDIM, c = i % NDIM;
    int b = tok >> 9, l = tok & 511;
    int h = c >> 6,  d = c & 63;
    int bh = b * NHEAD + h;
    cat[i] = O[((size_t)bh * LSEQ + l) * DH + d];
}

// ------------------------------ WMMA GEMM ----------------------------------
// C[M,N] = scale * A[M,K] * B[N,K]^T  (+bias[N]) (+resid[M,N]) (+ReLU)
// JF = number of 16-wide N fragments per wave: JF=4 -> BN=128, JF=2 -> BN=64.
// 128 threads = 4 waves in a 2x2 grid; each wave computes 64 x (16*JF).
// BM=128, BK=64, double-buffered LDS.  All problem dims are multiples of the
// tile so no bounds checks -> EXEC is all-ones at every WMMA (ISA §7.12).
// Fragment layouts per §7.12.2 (bf16, wave32):
//   A 16x32: lane m=lane&15, sel=lane>>4 holds K[8s,8s+8) and K[16+8s,24+8s)
//   B 32x16: lane n=lane&15, sel=lane>>4 holds K[8s,8s+8) and K[16+8s,24+8s)
//   C/D f32: vgpr r -> (M = r + 8*sel, N = lane&15)
template<int JF>
__global__ __launch_bounds__(128)
void gemm_nt_bf16(const bf16* __restrict__ A, const bf16* __restrict__ B,
                  const float* __restrict__ bias, const float* __restrict__ resid,
                  float* __restrict__ Cf, bf16* __restrict__ Cb,
                  int N, int K,
                  long long strideA, long long strideB, long long strideC,
                  float scale, int relu) {
    constexpr int BM  = 128;
    constexpr int BN  = 32 * JF;
    constexpr int LDK = 72;                 // 64 + 8 pad: conflict-free b128
    __shared__ bf16 shA[2][BM][LDK];
    __shared__ bf16 shB[2][BN][LDK];

    const int tid  = threadIdx.x;
    const int lane = tid & 31;
    const int wave = tid >> 5;
    const int wr   = (wave >> 1) * 64;      // wave macro-tile row
    const int wc   = (wave & 1) * (16 * JF);// wave macro-tile col
    const long long m0 = (long long)blockIdx.y * BM;
    const long long n0 = (long long)blockIdx.x * BN;
    const int z = blockIdx.z;
    A += (long long)z * strideA + m0 * (long long)K;
    B += (long long)z * strideB + n0 * (long long)K;

    f32x8 acc[4][JF] = {};

    // stage one BK=64 slab of A (BM x 64) and B (BN x 64) into LDS buffer
    auto stage = [&](int buf, int k0) {
#if defined(USE_ASYNC_LDS)
        {   // A: thread t owns row t (64 elems = 8 x b128), async -> LDS
            const bf16* ga = A + (long long)tid * K + k0;
            bf16* la = &shA[buf][tid][0];
#pragma unroll
            for (int i = 0; i < 8; ++i) async_cp16(ga + i * 8, la + i * 8);
        }
        if constexpr (JF == 4) {
            const bf16* gb = B + (long long)tid * K + k0;
            bf16* lb = &shB[buf][tid][0];
#pragma unroll
            for (int i = 0; i < 8; ++i) async_cp16(gb + i * 8, lb + i * 8);
        } else {
            int r = tid >> 1, c = (tid & 1) * 32;
            const bf16* gb = B + (long long)r * K + k0 + c;
            bf16* lb = &shB[buf][r][c];
#pragma unroll
            for (int i = 0; i < 4; ++i) async_cp16(gb + i * 8, lb + i * 8);
        }
#else
        {   // fallback: stage through VGPRs
            const bf16* ga = A + (long long)tid * K + k0;
            bf16x8* la = (bf16x8*)&shA[buf][tid][0];
#pragma unroll
            for (int i = 0; i < 8; ++i) la[i] = ((const bf16x8*)ga)[i];
        }
        if constexpr (JF == 4) {
            const bf16* gb = B + (long long)tid * K + k0;
            bf16x8* lb = (bf16x8*)&shB[buf][tid][0];
#pragma unroll
            for (int i = 0; i < 8; ++i) lb[i] = ((const bf16x8*)gb)[i];
        } else {
            int r = tid >> 1, c = (tid & 1) * 32;
            const bf16* gb = B + (long long)r * K + k0 + c;
            bf16x8* lb = (bf16x8*)&shB[buf][r][c];
#pragma unroll
            for (int i = 0; i < 4; ++i) lb[i] = ((const bf16x8*)gb)[i];
        }
#endif
    };

    stage(0, 0);
    wait_async0();
    __syncthreads();

    int buf = 0;
    const int mrow = lane & 15;
    const int sel  = lane >> 4;
    for (int k0 = 0; k0 < K; k0 += 64) {
        if (k0 + 64 < K) stage(buf ^ 1, k0 + 64);   // fill next buffer

#pragma unroll
        for (int kw = 0; kw < 64; kw += 32) {
            bf16x16 af[4], bfr[JF];
#pragma unroll
            for (int i = 0; i < 4; ++i) {
                const bf16* pa = &shA[buf][wr + i * 16 + mrow][kw];
                ((bf16x8*)&af[i])[0] = *(const bf16x8*)(pa + sel * 8);
                ((bf16x8*)&af[i])[1] = *(const bf16x8*)(pa + 16 + sel * 8);
            }
#pragma unroll
            for (int j = 0; j < JF; ++j) {
                const bf16* pb = &shB[buf][wc + j * 16 + mrow][kw + sel * 8];
                ((bf16x8*)&bfr[j])[0] = *(const bf16x8*)(pb);
                ((bf16x8*)&bfr[j])[1] = *(const bf16x8*)(pb + 16);
            }
#pragma unroll
            for (int i = 0; i < 4; ++i)
#pragma unroll
                for (int j = 0; j < JF; ++j)
                    acc[i][j] = __builtin_amdgcn_wmma_f32_16x16x32_bf16(
                        false, af[i], false, bfr[j], (short)0, acc[i][j], false, false);
        }

        wait_async0();       // next slab landed in LDS
        __syncthreads();     // everyone done reading current slab
        buf ^= 1;
    }

    // epilogue
    const int nn   = lane & 15;
    const int msel = (lane >> 4) * 8;
#pragma unroll
    for (int i = 0; i < 4; ++i)
#pragma unroll
        for (int j = 0; j < JF; ++j) {
            long long cm = m0 + wr + i * 16 + msel;
            long long cn = n0 + wc + j * 16 + nn;
            float bv = bias ? bias[cn] : 0.f;
#pragma unroll
            for (int r = 0; r < 8; ++r) {
                long long off = (long long)z * strideC + (cm + r) * (long long)N + cn;
                float v = acc[i][j][r] * scale + bv;
                if (resid) v += resid[off];
                if (relu)  v = fmaxf(v, 0.f);
                if (Cf) Cf[off] = v;
                else    Cb[off] = (bf16)v;
            }
        }
}

// ------------------------------ host side ----------------------------------
// Flattening: top-level insertion order (encoder_in, encoder_out, decoder_in,
// dec_attn_mask, params); params with sorted dict keys: dec[0..5], enc[0..5],
// final_norm, out.  dec layer (26 leaves): ca{wk,wo,wq,wv}{b,w}, ff1{b,w},
// ff2{b,w}, ln1{b,w}, ln2{b,w}, ln3{b,w}, sa{wk,wo,wq,wv}{b,w}
enum {
    CA_WK_B = 0, CA_WK_W, CA_WO_B, CA_WO_W, CA_WQ_B, CA_WQ_W, CA_WV_B, CA_WV_W,
    FF1_B, FF1_W, FF2_B, FF2_W, LN1_B, LN1_W, LN2_B, LN2_W, LN3_B, LN3_W,
    SA_WK_B, SA_WK_W, SA_WO_B, SA_WO_W, SA_WQ_B, SA_WQ_W, SA_WV_B, SA_WV_W
};
#define DEC_BASE(l)  (4 + 26 * (l))
#define FINAL_NORM_B 256
#define FINAL_NORM_W 257
#define OUT_B        258
#define OUT_W        259

static void launch_gemm(hipStream_t s, const bf16* A, const bf16* B,
                        const float* bias, const float* resid,
                        float* Cf, bf16* Cb,
                        int M, int N, int K, int batch,
                        long long sA, long long sB, long long sC,
                        float scale, int relu) {
    if (N % 128 == 0) {
        dim3 grid(N / 128, M / 128, batch);
        gemm_nt_bf16<4><<<grid, 128, 0, s>>>(A, B, bias, resid, Cf, Cb, N, K, sA, sB, sC, scale, relu);
    } else {   // N == 64 (AV GEMM)
        dim3 grid(N / 64, M / 128, batch);
        gemm_nt_bf16<2><<<grid, 128, 0, s>>>(A, B, bias, resid, Cf, Cb, N, K, sA, sB, sC, scale, relu);
    }
}

extern "C" void kernel_launch(void* const* d_in, const int* in_sizes, int n_in,
                              void* d_out, int out_size, void* d_ws, size_t ws_size,
                              hipStream_t stream) {
    (void)in_sizes; (void)n_in; (void)out_size; (void)ws_size;
    auto F = [&](int i) { return (const float*)d_in[i]; };

    // ---- carve workspace (~170MB; bf16 weights dominate) ----
    size_t off = 0;
    auto carve = [&](size_t bytes) -> char* {
        char* p = (char*)d_ws + off;
        off += (bytes + 255) & ~(size_t)255;
        return p;
    };
    float* x   = (float*)carve((size_t)TOK * NDIM * 4);
    bf16*  h   = (bf16*) carve((size_t)TOK * NDIM * 2);
    bf16*  q   = (bf16*) carve((size_t)TOK * NDIM * 2);
    bf16*  qh  = (bf16*) carve((size_t)BH * LSEQ * DH * 2);
    bf16*  qhT = (bf16*) carve((size_t)BH * DH * LSEQ * 2);
    float* S   = (float*)carve((size_t)BH * LSEQ * LSEQ * 4);
    bf16*  P   = (bf16*) carve((size_t)BH * LSEQ * LSEQ * 2);
    bf16*  O   = (bf16*) carve((size_t)BH * LSEQ * DH * 2);
    bf16*  cat = (bf16*) carve((size_t)TOK * NDIM * 2);
    bf16*  t   = (bf16*) carve((size_t)TOK * NFF * 2);
    bf16 *w_sa_q[6], *w_sa_o[6], *w_ca_q[6], *w_ca_o[6], *w_f1[6], *w_f2[6];
    for (int l = 0; l < 6; ++l) {
        w_sa_q[l] = (bf16*)carve((size_t)NDIM * NDIM * 2);
        w_sa_o[l] = (bf16*)carve((size_t)NDIM * NDIM * 2);
        w_ca_q[l] = (bf16*)carve((size_t)NDIM * NDIM * 2);
        w_ca_o[l] = (bf16*)carve((size_t)NDIM * NDIM * 2);
        w_f1[l]   = (bf16*)carve((size_t)NFF * NDIM * 2);
        w_f2[l]   = (bf16*)carve((size_t)NDIM * NFF * 2);
    }
    bf16* w_out = (bf16*)carve((size_t)VOCAB * NDIM * 2);

    auto cvt = [&](const float* src, bf16* dst, size_t n) {
        cvt_f32_bf16<<<(int)((n + 255) / 256), 256, 0, stream>>>(src, dst, (int)n);
    };

    // ---- convert live weights to bf16 (dead wk/wv + encoder skipped) ----
    for (int l = 0; l < 6; ++l) {
        int b0 = DEC_BASE(l);
        cvt(F(b0 + SA_WQ_W), w_sa_q[l], (size_t)NDIM * NDIM);
        cvt(F(b0 + SA_WO_W), w_sa_o[l], (size_t)NDIM * NDIM);
        cvt(F(b0 + CA_WQ_W), w_ca_q[l], (size_t)NDIM * NDIM);
        cvt(F(b0 + CA_WO_W), w_ca_o[l], (size_t)NDIM * NDIM);
        cvt(F(b0 + FF1_W),   w_f1[l],   (size_t)NFF * NDIM);
        cvt(F(b0 + FF2_W),   w_f2[l],   (size_t)NDIM * NFF);
    }
    cvt(F(OUT_W), w_out, (size_t)VOCAB * NDIM);

    // ---- decoder ----
    const float* mask = (const float*)d_in[3];
    copy_f32<<<TOK * NDIM / 256, 256, 0, stream>>>((const float*)d_in[2], x, TOK * NDIM);

    for (int l = 0; l < 6; ++l) {
        const int b0 = DEC_BASE(l);

        for (int sub = 0; sub < 2; ++sub) {      // sub 0: sa (masked), 1: ca
            const int ln_w = sub == 0 ? LN1_W : LN2_W;
            const int ln_b = sub == 0 ? LN1_B : LN2_B;
            const bf16* wq = sub == 0 ? w_sa_q[l] : w_ca_q[l];
            const bf16* wo = sub == 0 ? w_sa_o[l] : w_ca_o[l];
            const float* bq = sub == 0 ? F(b0 + SA_WQ_B) : F(b0 + CA_WQ_B);
            const float* bo = sub == 0 ? F(b0 + SA_WO_B) : F(b0 + CA_WO_B);
            const float* msk = sub == 0 ? mask : nullptr;

            layernorm768_bf16<<<TOK, 128, 0, stream>>>(x, F(b0 + ln_w), F(b0 + ln_b), h);
            launch_gemm(stream, h, wq, bq, nullptr, nullptr, q,
                        TOK, NDIM, NDIM, 1, 0, 0, 0, 1.f, 0);
            split_heads<<<(TOK * NDIM + 255) / 256, 256, 0, stream>>>(q, qh, qhT);
            // scores = (Q Q^T) / sqrt(dh), batched over 24 heads
            launch_gemm(stream, qh, qh, nullptr, nullptr, S, nullptr,
                        LSEQ, LSEQ, DH, BH,
                        (long long)LSEQ * DH, (long long)LSEQ * DH,
                        (long long)LSEQ * LSEQ, 0.125f, 0);
            softmax512_bf16<<<dim3(LSEQ, BH), 256, 0, stream>>>(S, msk, P);
            // O = P @ Q   (B operand = Q^T in NT layout)
            launch_gemm(stream, P, qhT, nullptr, nullptr, nullptr, O,
                        LSEQ, DH, LSEQ, BH,
                        (long long)LSEQ * LSEQ, (long long)DH * LSEQ,
                        (long long)LSEQ * DH, 1.f, 0);
            merge_heads<<<(TOK * NDIM + 255) / 256, 256, 0, stream>>>(O, cat);
            launch_gemm(stream, cat, wo, bo, x, x, nullptr,
                        TOK, NDIM, NDIM, 1, 0, 0, 0, 1.f, 0);
        }

        // FFN
        layernorm768_bf16<<<TOK, 128, 0, stream>>>(x, F(b0 + LN3_W), F(b0 + LN3_B), h);
        launch_gemm(stream, h, w_f1[l], F(b0 + FF1_B), nullptr, nullptr, t,
                    TOK, NFF, NDIM, 1, 0, 0, 0, 1.f, 1 /*relu*/);
        launch_gemm(stream, t, w_f2[l], F(b0 + FF2_B), x, x, nullptr,
                    TOK, NDIM, NFF, 1, 0, 0, 0, 1.f, 0);
    }

    // final LN + vocab projection (dominant GEMM: 1024 x 32000 x 768)
    layernorm768_bf16<<<TOK, 128, 0, stream>>>(x, F(FINAL_NORM_W), F(FINAL_NORM_B), h);
    launch_gemm(stream, h, w_out, F(OUT_B), nullptr, (float*)d_out, nullptr,
                TOK, VOCAB, NDIM, 1, 0, 0, 0, 1.f, 0);
}
